// LoRAConnector_39659728011343
// MI455X (gfx1250) — matrix operand order repaired and checked by reading, not compiled, not run on hardware
//
#include <hip/hip_runtime.h>

#define A_  8
#define B_  4
#define T_  1024
#define N_  4
#define C_  2048
#define F_  8192            // N*C
#define EPS_ 1e-5f
#define SINK_ITERS 20
#define TILE 16
#define PHIT_ROWS 32        // 16 res + 4 pre + 4 post + 8 zero pad

typedef __attribute__((ext_vector_type(16))) __bf16 v16bf;
typedef __attribute__((ext_vector_type(8)))  float  v8f;

// ---------------------------------------------------------------------------
// Kernel 0: pack Phi = [phi_res | phi_pre | phi_post | 0pad] into bf16,
// column-major-per-output (row r of phiT holds logit-column r over all f),
// so WMMA B fragments are contiguous 32B per lane.
// phiT layout: [A][32][F] bf16  (4 MB in d_ws)
// ---------------------------------------------------------------------------
__global__ void pack_phi_kernel(const float* __restrict__ phi_pre,
                                const float* __restrict__ phi_post,
                                const float* __restrict__ phi_res,
                                __bf16* __restrict__ phiT) {
  const int a = blockIdx.x >> 5;
  const int r = blockIdx.x & 31;
  for (int f = threadIdx.x; f < F_; f += blockDim.x) {
    float v = 0.0f;
    if (r < 16)      v = phi_res [((size_t)a * F_ + f) * 16 + r];
    else if (r < 20) v = phi_pre [((size_t)a * F_ + f) * 4  + (r - 16)];
    else if (r < 24) v = phi_post[((size_t)a * F_ + f) * 4  + (r - 20)];
    phiT[((size_t)(a * PHIT_ROWS + r)) * F_ + f] = (__bf16)v;
  }
}

// ---------------------------------------------------------------------------
// Kernel 1: fused per-16-token-tile kernel.
//   phase 1: 8 waves K-split WMMA -> 16x32 logits tile (+ sum-of-squares)
//   phase 2: 16 lanes: sigmoid gates + 4x4 Sinkhorn (registers)
//   phase 3: block-wide float4 streaming of x_agg and out
// ---------------------------------------------------------------------------
__global__ void __launch_bounds__(256)
lora_fused_kernel(const float* __restrict__ x,
                  const float* __restrict__ outp,
                  const float* __restrict__ w_all,
                  const float* __restrict__ b_pre,
                  const float* __restrict__ b_post,
                  const float* __restrict__ b_res,
                  const float* __restrict__ alpha_pre,
                  const float* __restrict__ alpha_post,
                  const float* __restrict__ alpha_res,
                  const int*   __restrict__ adapter_idx,
                  const __bf16* __restrict__ phiT,
                  float* __restrict__ agg_out,
                  float* __restrict__ main_out) {
  __shared__ float sm_logits[TILE][PHIT_ROWS];
  __shared__ float sm_ss[TILE];
  __shared__ float sm_M[TILE][16];
  __shared__ float sm_hpre[TILE][4];
  __shared__ float sm_hpost[TILE][4];

  const int tid  = threadIdx.x;
  const int tile = blockIdx.x;
  const int b    = tile >> 6;          // 64 tiles per batch (T/16)
  const int t0   = (tile & 63) << 4;
  const int aidx = adapter_idx[b];

  for (int i = tid; i < TILE * PHIT_ROWS; i += 256) ((float*)sm_logits)[i] = 0.0f;
  if (tid < TILE) sm_ss[tid] = 0.0f;
  __syncthreads();

  const int wid  = tid >> 5;
  const int lane = tid & 31;
  const int mcol = lane & 15;          // token row (A) / logit col (B/C/D)
  const int hf   = lane >> 4;          // lane half

  const size_t tok0 = (size_t)(b * T_ + t0);
  const float* xrow = x + (tok0 + mcol) * (size_t)F_;
  const float* wrow = w_all + (size_t)aidx * F_;
  const __bf16* brow0 = phiT + ((size_t)(aidx * PHIT_ROWS) + mcol)      * F_;
  const __bf16* brow1 = phiT + ((size_t)(aidx * PHIT_ROWS) + 16 + mcol) * F_;

  v8f acc0 = {};
  v8f acc1 = {};
  float ss = 0.0f;
  const int kbase = wid * (F_ / 8);    // 1024 per wave

  for (int it = 0; it < (F_ / 8) / 32; ++it) {   // 32 K-steps of 32
    const int kk = kbase + it * 32;
    const int o0 = kk + hf * 16;       // K block for VGPR0-3 of A frag
    const int o1 = kk + hf * 16 + 8;   // K block for VGPR4-7 of A frag

    float4 xa = *(const float4*)(xrow + o0);
    float4 xb = *(const float4*)(xrow + o0 + 4);
    float4 xc = *(const float4*)(xrow + o1);
    float4 xd = *(const float4*)(xrow + o1 + 4);
    float4 wa = *(const float4*)(wrow + o0);
    float4 wb = *(const float4*)(wrow + o0 + 4);
    float4 wc = *(const float4*)(wrow + o1);
    float4 wd = *(const float4*)(wrow + o1 + 4);

    // raw-x sum of squares for RMS (each element touched exactly once)
    ss += xa.x*xa.x + xa.y*xa.y + xa.z*xa.z + xa.w*xa.w;
    ss += xb.x*xb.x + xb.y*xb.y + xb.z*xb.z + xb.w*xb.w;
    ss += xc.x*xc.x + xc.y*xc.y + xc.z*xc.z + xc.w*xc.w;
    ss += xd.x*xd.x + xd.y*xd.y + xd.z*xd.z + xd.w*xd.w;

    v16bf av;
    av[0]  = (__bf16)(xa.x * wa.x);  av[1]  = (__bf16)(xa.y * wa.y);
    av[2]  = (__bf16)(xa.z * wa.z);  av[3]  = (__bf16)(xa.w * wa.w);
    av[4]  = (__bf16)(xb.x * wb.x);  av[5]  = (__bf16)(xb.y * wb.y);
    av[6]  = (__bf16)(xb.z * wb.z);  av[7]  = (__bf16)(xb.w * wb.w);
    av[8]  = (__bf16)(xc.x * wc.x);  av[9]  = (__bf16)(xc.y * wc.y);
    av[10] = (__bf16)(xc.z * wc.z);  av[11] = (__bf16)(xc.w * wc.w);
    av[12] = (__bf16)(xd.x * wd.x);  av[13] = (__bf16)(xd.y * wd.y);
    av[14] = (__bf16)(xd.z * wd.z);  av[15] = (__bf16)(xd.w * wd.w);

    v16bf bv0 = *(const v16bf*)(brow0 + kk + hf * 16);
    v16bf bv1 = *(const v16bf*)(brow1 + kk + hf * 16);

    acc0 = __builtin_amdgcn_wmma_f32_16x16x32_bf16(false, av, false, bv0,
                                                   (short)0, acc0, false, false);
    acc1 = __builtin_amdgcn_wmma_f32_16x16x32_bf16(false, av, false, bv1,
                                                   (short)0, acc1, false, false);
  }

  // reduce 8 partial 16x32 tiles into LDS (ds_add_f32)
#pragma unroll
  for (int r = 0; r < 8; ++r) {
    const int m = r + hf * 8;                  // C/D layout: lanes 16-31 -> M+8
    atomicAdd(&sm_logits[m][mcol],      acc0[r]);
    atomicAdd(&sm_logits[m][16 + mcol], acc1[r]);
  }
  atomicAdd(&sm_ss[mcol], ss);
  __syncthreads();

  // gates + Sinkhorn: one token per lane, all in registers
  if (tid < TILE) {
    const int i = tid;
    const float inv_rms = rsqrtf(sm_ss[i] * (1.0f / (float)F_) + EPS_);
    const float ap  = alpha_pre[aidx];
    const float apo = alpha_post[aidx];
    const float ar  = alpha_res[aidx];
#pragma unroll
    for (int n = 0; n < 4; ++n) {
      float zl = ap  * sm_logits[i][16 + n] * inv_rms + b_pre [aidx * 4 + n];
      float zp = apo * sm_logits[i][20 + n] * inv_rms + b_post[aidx * 4 + n];
      sm_hpre [i][n] = 1.0f / (1.0f + __expf(-zl));
      sm_hpost[i][n] = 2.0f / (1.0f + __expf(-zp));
    }
    float Mm[16];
#pragma unroll
    for (int m = 0; m < 16; ++m) {
      float z = ar * sm_logits[i][m] * inv_rms + b_res[aidx * 16 + m];
      Mm[m] = __expf(z);
    }
    for (int itr = 0; itr < SINK_ITERS; ++itr) {
#pragma unroll
      for (int rr = 0; rr < 4; ++rr) {
        float s = Mm[4*rr] + Mm[4*rr+1] + Mm[4*rr+2] + Mm[4*rr+3];
        float inv = 1.0f / s;
        Mm[4*rr] *= inv; Mm[4*rr+1] *= inv; Mm[4*rr+2] *= inv; Mm[4*rr+3] *= inv;
      }
#pragma unroll
      for (int cc = 0; cc < 4; ++cc) {
        float s = Mm[cc] + Mm[4+cc] + Mm[8+cc] + Mm[12+cc];
        float inv = 1.0f / s;
        Mm[cc] *= inv; Mm[4+cc] *= inv; Mm[8+cc] *= inv; Mm[12+cc] *= inv;
      }
    }
#pragma unroll
    for (int m = 0; m < 16; ++m) sm_M[i][m] = Mm[m];
  }
  __syncthreads();

  // streaming output phase (bandwidth-bound part)
  const float4* x4   = (const float4*)(x        + tok0 * (size_t)F_);
  const float4* o4   = (const float4*)(outp     + tok0 * (size_t)C_);
  float4*       agg4 = (float4*)      (agg_out  + tok0 * (size_t)C_);
  float4*       out4 = (float4*)      (main_out + tok0 * (size_t)F_);
  const int C4 = C_ / 4;   // 512

  for (int idx = tid; idx < TILE * C4; idx += 256) {
    const int i  = idx >> 9;
    const int c4 = idx & (C4 - 1);
    const size_t xb0 = (size_t)i * (4 * C4) + c4;

    __builtin_prefetch(&x4[xb0 + 2 * C4 * 4], 0, 1);   // global_prefetch hint

    float4 xv0 = x4[xb0 + 0 * C4];
    float4 xv1 = x4[xb0 + 1 * C4];
    float4 xv2 = x4[xb0 + 2 * C4];
    float4 xv3 = x4[xb0 + 3 * C4];
    float4 ov  = o4[(size_t)i * C4 + c4];

    const float h0 = sm_hpre[i][0], h1 = sm_hpre[i][1];
    const float h2 = sm_hpre[i][2], h3 = sm_hpre[i][3];
    float4 ag;
    ag.x = h0*xv0.x + h1*xv1.x + h2*xv2.x + h3*xv3.x;
    ag.y = h0*xv0.y + h1*xv1.y + h2*xv2.y + h3*xv3.y;
    ag.z = h0*xv0.z + h1*xv1.z + h2*xv2.z + h3*xv3.z;
    ag.w = h0*xv0.w + h1*xv1.w + h2*xv2.w + h3*xv3.w;
    agg4[(size_t)i * C4 + c4] = ag;

#pragma unroll
    for (int n = 0; n < 4; ++n) {
      const float m0 = sm_M[i][n*4+0], m1 = sm_M[i][n*4+1];
      const float m2 = sm_M[i][n*4+2], m3 = sm_M[i][n*4+3];
      const float hq = sm_hpost[i][n];
      float4 r;
      r.x = m0*xv0.x + m1*xv1.x + m2*xv2.x + m3*xv3.x + hq*ov.x;
      r.y = m0*xv0.y + m1*xv1.y + m2*xv2.y + m3*xv3.y + hq*ov.y;
      r.z = m0*xv0.z + m1*xv1.z + m2*xv2.z + m3*xv3.z + hq*ov.z;
      r.w = m0*xv0.w + m1*xv1.w + m2*xv2.w + m3*xv3.w + hq*ov.w;
      out4[(size_t)(i * 4 + n) * C4 + c4] = r;
    }
  }
}

// ---------------------------------------------------------------------------
extern "C" void kernel_launch(void* const* d_in, const int* in_sizes, int n_in,
                              void* d_out, int out_size, void* d_ws, size_t ws_size,
                              hipStream_t stream) {
  (void)in_sizes; (void)n_in; (void)out_size; (void)ws_size;

  const float* x        = (const float*)d_in[0];
  const float* outp     = (const float*)d_in[1];
  const float* w_all    = (const float*)d_in[2];
  const float* phi_pre  = (const float*)d_in[3];
  const float* phi_post = (const float*)d_in[4];
  const float* phi_res  = (const float*)d_in[5];
  const float* b_pre    = (const float*)d_in[6];
  const float* b_post   = (const float*)d_in[7];
  const float* b_res    = (const float*)d_in[8];
  const float* a_pre    = (const float*)d_in[9];
  const float* a_post   = (const float*)d_in[10];
  const float* a_res    = (const float*)d_in[11];
  const int*   aidx     = (const int*)d_in[12];

  float* agg = (float*)d_out;                                   // (B,T,C)
  float* mo  = (float*)d_out + (size_t)B_ * T_ * C_;            // (B,T,N,C)
  __bf16* phiT = (__bf16*)d_ws;                                 // 4 MB

  pack_phi_kernel<<<A_ * PHIT_ROWS, 256, 0, stream>>>(phi_pre, phi_post, phi_res, phiT);
  lora_fused_kernel<<<(B_ * T_) / TILE, 256, 0, stream>>>(
      x, outp, w_all, b_pre, b_post, b_res, a_pre, a_post, a_res,
      aidx, phiT, agg, mo);
}
